// W4A8B8O8Linear_81913616270133
// MI455X (gfx1250) — 8-bit matrix operand order repaired and checked by reading, compiled, not run on hardware
//
#include <hip/hip_runtime.h>

typedef int v8i  __attribute__((ext_vector_type(8)));
typedef int v4i_ __attribute__((ext_vector_type(4)));
typedef int v2i_ __attribute__((ext_vector_type(2)));

#define TOKENS 8192
#define KDIM   4096
#define NDIM   4096
#define GRP    128
#define NGRP   (KDIM / GRP)

#define BM 128
#define BN 256
#define BK 64
#define NKSTEPS (KDIM / BK)
#define LDS_STRIDE 80

__global__ void pack_x_kernel(const int* __restrict__ x32, char* __restrict__ x8, int n) {
  int i = (blockIdx.x * blockDim.x + threadIdx.x) * 16;
  if (i >= n) return;
  const v4i_* src = (const v4i_*)(x32 + i);
  int dw[4] __attribute__((aligned(16)));
#pragma unroll
  for (int q = 0; q < 4; ++q) {
    v4i_ v = src[q];
    dw[q] = (v.x & 0xFF) | ((v.y & 0xFF) << 8) | ((v.z & 0xFF) << 16) | ((int)((unsigned)v.w << 24));
  }
  *(v4i_*)(x8 + i) = *(const v4i_*)dw;
}

__global__ void dequant_w_kernel(const int* __restrict__ wpk,
                                 const int* __restrict__ scales,
                                 const int* __restrict__ zeros,
                                 char* __restrict__ w8) {
  const int t = blockIdx.x * blockDim.x + threadIdx.x;
  const int pk_per_row  = KDIM / 2;
  const int thr_per_row = pk_per_row / 8;
  const int o  = t / thr_per_row;
  const int jp = (t % thr_per_row) * 8;
  const int k0 = jp * 2;
  const int g  = k0 / GRP;
  const int z  = zeros [o * NGRP + g];
  const int s  = scales[o * NGRP + g];
  const v4i_* src = (const v4i_*)(wpk + (size_t)o * pk_per_row + jp);
  v4i_ p0 = src[0], p1 = src[1];
  int pv[8] = {p0.x, p0.y, p0.z, p0.w, p1.x, p1.y, p1.z, p1.w};
  int dw[4] __attribute__((aligned(16)));
#pragma unroll
  for (int q = 0; q < 4; ++q) {
    int a0 = pv[2 * q], a1 = pv[2 * q + 1];
    int b0 = (( a0       & 0xF) - z) * s;
    int b1 = (((a0 >> 4) & 0xF) - z) * s;
    int b2 = (( a1       & 0xF) - z) * s;
    int b3 = (((a1 >> 4) & 0xF) - z) * s;
    dw[q] = (b0 & 0xFF) | ((b1 & 0xFF) << 8) | ((b2 & 0xFF) << 16) | ((int)((unsigned)(b3 & 0xFF) << 24));
  }
  *(v4i_*)(w8 + (size_t)o * KDIM + k0) = *(const v4i_*)dw;
}

__device__ __forceinline__ void async_b128(unsigned lds_off, const char* g) {
  asm volatile("global_load_async_to_lds_b128 %0, %1, off"
               :: "v"(lds_off), "v"((unsigned long long)(size_t)g)
               : "memory");
}
__device__ __forceinline__ void wait_async0() {
  asm volatile("s_wait_asynccnt 0" ::: "memory");
}

__launch_bounds__(256)
__global__ void w4a8_gemm_kernel(const char* __restrict__ x8,
                                 const char* __restrict__ w8,
                                 const float* __restrict__ a,
                                 const float* __restrict__ bscale,
                                 const int* __restrict__ bias,
                                 float* __restrict__ out) {
  __shared__ __align__(16) char As[2][BM * LDS_STRIDE];
  __shared__ __align__(16) char Bs[2][BN * LDS_STRIDE];

  const int tid  = threadIdx.x;
  const int lane = tid & 31;
  const int wave = tid >> 5;
  const int wm   = wave >> 2;
  const int wn   = wave & 3;
  const int m0   = blockIdx.y * BM;
  const int n0   = blockIdx.x * BN;
  const int lmod  = lane & 15;
  const int lhalf = lane >> 4;

  const int arow = tid >> 1;
  const int acol = (tid & 1) * 32;
  const char* gA = x8 + (size_t)(m0 + arow) * KDIM + acol;
  const char* gB = w8 + (size_t)(n0 + tid) * KDIM;
  unsigned ldsA[2], ldsB[2];
  ldsA[0] = (unsigned)(size_t)&As[0][arow * LDS_STRIDE + acol];
  ldsA[1] = (unsigned)(size_t)&As[1][arow * LDS_STRIDE + acol];
  ldsB[0] = (unsigned)(size_t)&Bs[0][tid * LDS_STRIDE];
  ldsB[1] = (unsigned)(size_t)&Bs[1][tid * LDS_STRIDE];

  auto issue_tile = [&](int buf, int k0) {
    async_b128(ldsA[buf],      gA + k0);
    async_b128(ldsA[buf] + 16, gA + k0 + 16);
    async_b128(ldsB[buf],      gB + k0);
    async_b128(ldsB[buf] + 16, gB + k0 + 16);
    async_b128(ldsB[buf] + 32, gB + k0 + 32);
    async_b128(ldsB[buf] + 48, gB + k0 + 48);
  };

  v8i acc[4][4];
#pragma unroll
  for (int mf = 0; mf < 4; ++mf)
#pragma unroll
    for (int nf = 0; nf < 4; ++nf)
#pragma unroll
      for (int e = 0; e < 8; ++e) acc[mf][nf][e] = 0;

  auto compute_tile = [&](const char* Ab, const char* Bb) {
    v8i afrag[4];
#pragma unroll
    for (int mf = 0; mf < 4; ++mf) {
      const char* p = Ab + (wm * 64 + mf * 16 + lmod) * LDS_STRIDE + lhalf * 16;
      int ab[8] __attribute__((aligned(32)));
      *(v4i_*)(ab + 0) = *(const v4i_*)(p + 0);
      *(v4i_*)(ab + 4) = *(const v4i_*)(p + 32);
      afrag[mf] = *(const v8i*)ab;
    }
#pragma unroll
    for (int nf = 0; nf < 4; ++nf) {
      const char* p = Bb + (wn * 64 + nf * 16 + lmod) * LDS_STRIDE + lhalf * 16;
      int bb[8] __attribute__((aligned(32)));
      *(v4i_*)(bb + 0) = *(const v4i_*)(p + 0);
      *(v4i_*)(bb + 4) = *(const v4i_*)(p + 32);
      v8i bfrag = *(const v8i*)bb;
#pragma unroll
      for (int mf = 0; mf < 4; ++mf)
        acc[mf][nf] = __builtin_amdgcn_wmma_i32_16x16x64_iu8(
             true, afrag[mf],  true, bfrag,
            acc[mf][nf],  false,  false);
    }
  };

  issue_tile(0, 0);
  for (int kt = 0; kt < NKSTEPS; kt += 2) {
    wait_async0();
    __syncthreads();
    issue_tile(1, (kt + 1) * BK);
    compute_tile(As[0], Bs[0]);

    wait_async0();
    __syncthreads();
    if (kt + 2 < NKSTEPS) issue_tile(0, (kt + 2) * BK);
    compute_tile(As[1], Bs[1]);
  }

  const float bs = bscale[0];
#pragma unroll
  for (int nf = 0; nf < 4; ++nf) {
    const int n = n0 + wn * 64 + nf * 16 + lmod;
    const float fa = a[n];
    const float fb = bs * (float)bias[n];
#pragma unroll
    for (int mf = 0; mf < 4; ++mf) {
      const int mbase = m0 + wm * 64 + mf * 16 + lhalf * 8;
#pragma unroll
      for (int r = 0; r < 8; ++r) {
        float v = __builtin_rintf(fa * (float)acc[mf][nf][r] + fb);
        v = fminf(fmaxf(v, -128.0f), 127.0f);
        out[(size_t)(mbase + r) * NDIM + n] = v;
      }
    }
  }
}

extern "C" void kernel_launch(void* const* d_in, const int* in_sizes, int n_in,
                              void* d_out, int out_size, void* d_ws, size_t ws_size,
                              hipStream_t stream) {
  const int*   x32    = (const int*)d_in[0];
  const int*   wpk    = (const int*)d_in[1];
  const int*   bias   = (const int*)d_in[2];
  const float* a      = (const float*)d_in[3];
  const float* bsc    = (const float*)d_in[4];
  const int*   scales = (const int*)d_in[5];
  const int*   zeros  = (const int*)d_in[6];
  float*       out    = (float*)d_out;

  char* x8 = (char*)d_ws;
  char* w8 = (char*)d_ws + (size_t)TOKENS * KDIM;

  {
    const int n = TOKENS * KDIM;
    const int threads = n / 16;
    pack_x_kernel<<<threads / 256, 256, 0, stream>>>(x32, x8, n);
  }
  {
    const int threads = (NDIM * KDIM) / 16;
    dequant_w_kernel<<<threads / 256, 256, 0, stream>>>(wpk, scales, zeros, w8);
  }
  {
    dim3 grid(NDIM / BN, TOKENS / BM);
    w4a8_gemm_kernel<<<grid, 256, 0, stream>>>(x8, w8, a, bsc, bias, out);
  }
}
